// TFN_85538568667433
// MI455X (gfx1250) — matrix operand order repaired and checked by reading, not compiled, not run on hardware
//
#include <hip/hip_runtime.h>

// MI455X / gfx1250 fused Tensor-Field-Network layer.
// N=512 atoms, C=32 channels, R=64 RBFs. All heavy math on v_wmma_f32_16x16x32_f16.

typedef _Float16 v16h __attribute__((ext_vector_type(16)));
typedef _Float16 v8h  __attribute__((ext_vector_type(8)));
typedef float    v8f  __attribute__((ext_vector_type(8)));

#define NATOMS 512
#define NC     32
#define NR     64

// Raw hardware transcendentals (no libm range-fixup chains).
__device__ __forceinline__ float hw_exp2(float x) {
#if __has_builtin(__builtin_amdgcn_exp2f)
  return __builtin_amdgcn_exp2f(x);
#else
  float r; asm volatile("v_exp_f32 %0, %1" : "=v"(r) : "v"(x)); return r;
#endif
}
__device__ __forceinline__ float hw_sqrt(float x) {
#if __has_builtin(__builtin_amdgcn_sqrtf)
  return __builtin_amdgcn_sqrtf(x);
#else
  float r; asm volatile("v_sqrt_f32 %0, %1" : "=v"(r) : "v"(x)); return r;
#endif
}
__device__ __forceinline__ float hw_rsq(float x) {
#if __has_builtin(__builtin_amdgcn_rsqf)
  return __builtin_amdgcn_rsqf(x);
#else
  float r; asm volatile("v_rsq_f32 %0, %1" : "=v"(r) : "v"(x)); return r;
#endif
}

struct WPtrs {
  const float* w1[4];
  const float* b1[4];
  const float* w2[4];
  const float* b2[4];
};

__global__ __launch_bounds__(128)
void tfn_fused(const float* __restrict__ geom,
               const float* __restrict__ x0,
               const float* __restrict__ x1,
               WPtrs wp,
               float* __restrict__ out)
{
  const int tid  = threadIdx.x;
  const int lane = tid & 31;
  const int wave = tid >> 5;
  const int lrow = lane & 15;   // A-matrix row / B-matrix column / D-matrix column
  const int lhi  = lane >> 4;   // half-wave select
  const int a    = blockIdx.x;

  // Weights staged as f16 in WMMA B-matrix layout: tile[lane][e] with
  // k = ((e&8)?16:0) + (lane>>4)*8 + (e&7), n = lane&15; each lane's operand = 16 contiguous halves.
  __shared__ __align__(32) _Float16 w1t[4][2][4][512]; // [filter][kchunk][nchunk]
  __shared__ __align__(32) _Float16 w2t[4][2][2][512]; // [filter][kchunk][nchunk]
  __shared__ __align__(32) _Float16 Hs[4][16][64];     // per-wave hidden staging (row-major)
  __shared__ float oacc[32][12];                       // per-channel output accumulators

  // ---- stage weights into LDS (flat index t matches the array layout exactly) ----
  {
    _Float16* w1f = &w1t[0][0][0][0];
    for (int t = tid; t < 4*2*4*512; t += 128) {
      const int e  = t & 15;
      const int ln = (t >> 4) & 31;
      const int nc = (t >> 9) & 3;
      const int kc = (t >> 11) & 1;
      const int f  = (t >> 12) & 3;
      const int n  = nc*16 + (ln & 15);
      const int k  = kc*32 + ((e & 8) ? 16 : 0) + (ln >> 4)*8 + (e & 7);
      w1f[t] = (_Float16)wp.w1[f][n*NR + k];     // H[..,n] = sum_k rbf[..,k]*w1[n,k]
    }
    _Float16* w2f = &w2t[0][0][0][0];
    for (int t = tid; t < 4*2*2*512; t += 128) {
      const int e   = t & 15;
      const int ln  = (t >> 4) & 31;
      const int nc2 = (t >> 9) & 1;
      const int kc2 = (t >> 10) & 1;
      const int f   = (t >> 11) & 3;
      const int c   = nc2*16 + (ln & 15);
      const int h   = kc2*32 + ((e & 8) ? 16 : 0) + (ln >> 4)*8 + (e & 7);
      w2f[t] = (_Float16)wp.w2[f][c*NR + h];     // rad[..,c] = sum_h H[..,h]*w2[c,h]
    }
    for (int t = tid; t < 32*12; t += 128) (&oacc[0][0])[t] = 0.0f;
  }
  __syncthreads();

  // ---- per-lane bias values (added post-GEMM; C matrices start from inline 0) ----
  float b1v[4][4], b2v[4][2];
#pragma unroll
  for (int f = 0; f < 4; ++f) {
#pragma unroll
    for (int nc = 0; nc < 4; ++nc) b1v[f][nc] = wp.b1[f][nc*16 + lrow];
#pragma unroll
    for (int n2 = 0; n2 < 2; ++n2) b2v[f][n2] = wp.b2[f][n2*16 + lrow];
  }

  const float gax = geom[a*3+0];
  const float gay = geom[a*3+1];
  const float gaz = geom[a*3+2];

  // acc[nc2][slot]: 0=out0_a 1=out0_b 2..4=out1_a 5..7=out1_b 8..10=out1_c
  float acc[2][11];
#pragma unroll
  for (int i = 0; i < 2; ++i)
#pragma unroll
    for (int j = 0; j < 11; ++j) acc[i][j] = 0.0f;

  const float NL2E = -10.0f * 1.4426950408889634f;  // -gamma * log2(e), for v_exp_f32

  for (int bt = wave; bt < NATOMS/16; bt += 4) {
    const int bbase = bt*16;
    __builtin_prefetch(&x0[bbase*NC], 0, 1);
    __builtin_prefetch(&x1[bbase*NC*3], 0, 1);

    // ---- RBF A-matrix (16 pairs x 64 K), built in-register in A layout ----
    v16h A1[2];
    {
      const float* gA = geom + (bbase + lrow)*3;    // base + imm offsets
      const float dx = gax - gA[0];
      const float dy = gay - gA[1];
      const float dz = gaz - gA[2];
      const float r2 = dx*dx + dy*dy + dz*dz;
      const float d  = hw_sqrt(fmaxf(r2, 1e-8f));
#pragma unroll
      for (int kc = 0; kc < 2; ++kc)
#pragma unroll
        for (int e = 0; e < 16; ++e) {
          const int k = kc*32 + ((e & 8) ? 16 : 0) + lhi*8 + (e & 7);
          const float cen = (10.0f/63.0f) * (float)k;
          const float u = d - cen;
          A1[kc][e] = (_Float16)hw_exp2((NL2E * u) * u);  // bare v_exp_f32
        }
    }

    // ---- unit vectors for this lane's D-matrix rows (m = lhi*8 + v) ----
    float ux[8], uy[8], uz[8];
    {
      const float* gB = geom + (bbase + lhi*8)*3;   // base + imm offsets
#pragma unroll
      for (int v = 0; v < 8; ++v) {
        const float dx = gax - gB[v*3+0];
        const float dy = gay - gB[v*3+1];
        const float dz = gaz - gB[v*3+2];
        const float r2 = dx*dx + dy*dy + dz*dz;
        const float inv = hw_rsq(r2 + 1e-8f);       // == 0-producing for b==a (rij==0)
        ux[v] = dx*inv; uy[v] = dy*inv; uz[v] = dz*inv;
      }
    }

    // ---- four radial MLPs, 48 WMMAs per b-tile ----
    v8f rad[4][2];
#pragma unroll
    for (int f = 0; f < 4; ++f) {
      // layer 1: [16x64] = relu(rbf[16x64] x w1^T + b1)
#pragma unroll
      for (int nc = 0; nc < 4; ++nc) {
        const v8f zero = {};
        v8f h = __builtin_amdgcn_wmma_f32_16x16x32_f16(
                    false, A1[0], false, *(const v16h*)&w1t[f][0][nc][lane << 4],
                    (short)0, zero, false, false);          // inline SRC2 = 0
        h = __builtin_amdgcn_wmma_f32_16x16x32_f16(
                    false, A1[1], false, *(const v16h*)&w1t[f][1][nc][lane << 4],
                    (short)0, h, false, false);
        const float bb = b1v[f][nc];
#pragma unroll
        for (int v = 0; v < 8; ++v)
          Hs[wave][lhi*8 + v][nc*16 + lrow] = (_Float16)fmaxf(h[v] + bb, 0.0f);
      }
      asm volatile("s_wait_dscnt 0" ::: "memory");  // D-layout store -> A-layout load alias

      // reload H as layer-2 A-matrix: K runs of 8 halves -> ds_load_b128 pairs
      v16h A2[2];
#pragma unroll
      for (int kc = 0; kc < 2; ++kc) {
        const v8h lo = *(const v8h*)&Hs[wave][lrow][kc*32 + lhi*8];
        const v8h hi = *(const v8h*)&Hs[wave][lrow][kc*32 + lhi*8 + 16];
#pragma unroll
        for (int i = 0; i < 8; ++i) { A2[kc][i] = lo[i]; A2[kc][8+i] = hi[i]; }
      }

      // layer 2: rad[16x32] = H x w2^T + b2
#pragma unroll
      for (int n2 = 0; n2 < 2; ++n2) {
        const v8f zero = {};
        v8f r = __builtin_amdgcn_wmma_f32_16x16x32_f16(
                    false, A2[0], false, *(const v16h*)&w2t[f][0][n2][lane << 4],
                    (short)0, zero, false, false);          // inline SRC2 = 0
        r = __builtin_amdgcn_wmma_f32_16x16x32_f16(
                    false, A2[1], false, *(const v16h*)&w2t[f][1][n2][lane << 4],
                    (short)0, r, false, false);
        const float bb = b2v[f][n2];
#pragma unroll
        for (int v = 0; v < 8; ++v) r[v] += bb;
        rad[f][n2] = r;
      }
    }

    // ---- neighbor contraction on D-layout tiles (lane: c = n2*16+lrow, rows lhi*8+v) ----
#pragma unroll
    for (int n2 = 0; n2 < 2; ++n2) {
      const int c = n2*16 + lrow;
      const float* p0 = x0 + (bbase + lhi*8)*NC + c;        // base + imm offsets
      const float* p1 = x1 + ((bbase + lhi*8)*NC + c)*3;
#pragma unroll
      for (int v = 0; v < 8; ++v) {
        const float xv = p0[v*NC];
        const float xx = p1[v*NC*3 + 0];
        const float xy = p1[v*NC*3 + 1];
        const float xz = p1[v*NC*3 + 2];
        const float r0 = rad[0][n2][v];
        const float ra = rad[1][n2][v];
        const float rb = rad[2][n2][v];
        const float rc = rad[3][n2][v];
        acc[n2][0] += r0*xv;                               // out0_a
        acc[n2][5] += r0*xx; acc[n2][6] += r0*xy; acc[n2][7] += r0*xz;  // out1_b
        const float s = ra*xv;                             // out1_a = unit * rad * x0
        acc[n2][2] += s*ux[v]; acc[n2][3] += s*uy[v]; acc[n2][4] += s*uz[v];
        acc[n2][1] += rb*(ux[v]*xx + uy[v]*xy + uz[v]*xz); // out0_b = rad * (unit . x1)
        acc[n2][8]  += rc*(uy[v]*xz - uz[v]*xy);           // out1_c = rad * (unit x x1)
        acc[n2][9]  += rc*(uz[v]*xx - ux[v]*xz);
        acc[n2][10] += rc*(ux[v]*xy - uy[v]*xx);
      }
    }
  }

  // ---- block reduction over 4 waves x 2 half-lanes via LDS float atomics ----
#pragma unroll
  for (int n2 = 0; n2 < 2; ++n2) {
    const int c = n2*16 + lrow;
#pragma unroll
    for (int j = 0; j < 11; ++j) atomicAdd(&oacc[c][j], acc[n2][j]);
  }
  __syncthreads();

  if (tid < 32) {
    const int c = tid;
    out[a*64 + c]      = oacc[c][0];   // out0 bucket: [N,2C,1]
    out[a*64 + 32 + c] = oacc[c][1];
    float* o1 = out + NATOMS*64 + a*288;  // out1 bucket: [N,3C,3]
#pragma unroll
    for (int i = 0; i < 3; ++i) {
      o1[c*3 + i]      = oacc[c][2+i];
      o1[(32+c)*3 + i] = oacc[c][5+i];
      o1[(64+c)*3 + i] = oacc[c][8+i];
    }
  }
}

extern "C" void kernel_launch(void* const* d_in, const int* in_sizes, int n_in,
                              void* d_out, int out_size, void* d_ws, size_t ws_size,
                              hipStream_t stream) {
  (void)in_sizes; (void)n_in; (void)out_size; (void)d_ws; (void)ws_size;
  const float* geom = (const float*)d_in[0];
  const float* x0   = (const float*)d_in[1];
  const float* x1   = (const float*)d_in[2];
  WPtrs wp;
  for (int f = 0; f < 4; ++f) {
    wp.w1[f] = (const float*)d_in[3 + f*4 + 0];
    wp.b1[f] = (const float*)d_in[3 + f*4 + 1];
    wp.w2[f] = (const float*)d_in[3 + f*4 + 2];
    wp.b2[f] = (const float*)d_in[3 + f*4 + 3];
  }
  tfn_fused<<<NATOMS, 128, 0, stream>>>(geom, x0, x1, wp, (float*)d_out);
}
